// Net_90074054132250
// MI455X (gfx1250) — matrix operand order repaired and checked by reading, not compiled, not run on hardware
//
#include <hip/hip_runtime.h>
#include <stdint.h>

#define NMAX  4096
#define WMAX  128      // adjacency words per row at N=4096
#define HID   64       // padded feature stride
#define NHEAD 3

typedef __attribute__((ext_vector_type(16))) __bf16   v16bf;
typedef __attribute__((ext_vector_type(8)))  float    v8f;
typedef __attribute__((ext_vector_type(8)))  unsigned v8u;

__device__ __forceinline__ unsigned pk_bf16(float a, float b){
  __bf16 x=(__bf16)a, y=(__bf16)b;
  unsigned short ux=*(unsigned short*)&x, uy=*(unsigned short*)&y;
  return (unsigned)ux | ((unsigned)uy<<16);
}

// ---------------------------------------------------------------- utilities
__global__ void k_zero_f32(float* p, int n){ int i=blockIdx.x*blockDim.x+threadIdx.x; if(i<n) p[i]=0.f; }
__global__ void k_zero_u32(unsigned* p, int n){ int i=blockIdx.x*blockDim.x+threadIdx.x; if(i<n) p[i]=0u; }
__global__ void k_copy_f32(const float* s, float* d, int n){ int i=blockIdx.x*blockDim.x+threadIdx.x; if(i<n) d[i]=s[i]; }
__global__ void k_relu(float* p, int n){ int i=blockIdx.x*blockDim.x+threadIdx.x; if(i<n) p[i]=fmaxf(p[i],0.f); }

// ------------------------------------------------- bit-packed adjacency ops
__global__ void k_build_adj(const int* __restrict__ ei, int nE, unsigned* A, int n, int words){
  int e = blockIdx.x*blockDim.x+threadIdx.x;
  if(e < n) atomicOr(&A[(size_t)e*words + (e>>5)], 1u<<(e&31));   // forced diag
  if(e < nE){
    int r = ei[e], c = ei[nE+e];
    atomicOr(&A[(size_t)r*words + (c>>5)], 1u<<(c&31));
  }
}

__global__ void k_transpose_bits(const unsigned* __restrict__ A, unsigned* __restrict__ At, int n, int words){
  int idx = blockIdx.x*blockDim.x+threadIdx.x;
  if(idx >= n*words) return;
  int i = idx/words, wc = idx - i*words;
  unsigned w = 0u;
  for(int b=0;b<32;b++){
    int j = (wc<<5)+b;
    if(j<n) w |= ((A[(size_t)j*words + (i>>5)] >> (i&31)) & 1u) << b;
  }
  At[idx] = w;
}

// boolean A@A on bit rows: one block per output row, thread t owns word t
__global__ void k_sq_adj(const unsigned* __restrict__ A, unsigned* __restrict__ C, int n, int words){
  int i = blockIdx.x, t = threadIdx.x;
  unsigned acc = 0u;
  for(int w=0; w<words; w++){
    unsigned rw = A[(size_t)i*words + w];          // uniform scalar load
    while(rw){
      int b = __ffs(rw)-1; rw &= rw-1;
      int k = (w<<5)+b;
      if(t<words) acc |= A[(size_t)k*words + t];
    }
  }
  if(t<words) C[(size_t)i*words + t] = acc;
}

__global__ void k_gather_adj(const unsigned* __restrict__ A, const int* __restrict__ perm,
                             unsigned* __restrict__ B, int win, int nout, int wout){
  int idx = blockIdx.x*blockDim.x+threadIdx.x;
  if(idx >= nout*wout) return;
  int r = idx/wout, wc = idx - r*wout;
  int pi = perm[r];
  unsigned w = 0u;
  for(int b=0;b<32;b++){
    int c = (wc<<5)+b;
    if(c<nout){
      int pj = perm[c];
      w |= ((A[(size_t)pi*win + (pj>>5)] >> (pj&31)) & 1u) << b;
    }
  }
  B[idx] = w;
}

// -------------------------------- bf16 packing (zero-padded, fragment-friendly)
// features: [nrow16][kpadU] u32 of bf16 pairs (row-major, k contiguous)
__global__ void k_pack_feat(const float* __restrict__ x, int n, int fin, int kpadU,
                            unsigned* __restrict__ o, int nrow){
  int idx = blockIdx.x*blockDim.x+threadIdx.x;
  if(idx >= nrow*kpadU) return;
  int r = idx/kpadU, q = idx - r*kpadU;
  int k0 = q<<1;
  float a = (r<n && k0   < fin) ? x[(size_t)r*fin + k0    ] : 0.f;
  float b = (r<n && k0+1 < fin) ? x[(size_t)r*fin + k0 + 1] : 0.f;
  o[idx] = pk_bf16(a,b);
}

// W transposed+padded: [H][64 cols][32] u32 of bf16 k-pairs
__global__ void k_pack_w(const float* __restrict__ W, int fin, int fout, int H,
                         unsigned* __restrict__ o){
  int idx = blockIdx.x*blockDim.x+threadIdx.x;
  if(idx >= H*64*32) return;
  int h = idx>>11, rem = idx & 2047, col = rem>>5, q = rem&31, k0 = q<<1;
  float a = (k0   < fin && col < fout) ? W[((size_t)h*fin + k0    )*fout + col] : 0.f;
  float b = (k0+1 < fin && col < fout) ? W[((size_t)h*fin + k0 + 1)*fout + col] : 0.f;
  o[idx] = pk_bf16(a,b);
}

__global__ void k_pack_vec(const float* __restrict__ v, int fout, int H, float* __restrict__ o){
  int idx = blockIdx.x*blockDim.x+threadIdx.x;
  if(idx >= H*64) return;
  int h = idx>>6, c = idx&63;
  o[idx] = (c<fout) ? v[(size_t)h*fout + c] : 0.f;
}

// ------------------------------------ GAT feature transform: h = x@W, ss, sd
// grid(ceil(n/16), H), block 32. Fully unrolled bf16 WMMA, fp32 accumulate.
template<int KCH, int TILES>
__global__ void k_gat_xform(const unsigned* __restrict__ xbf, int kpadU,
                            const unsigned* __restrict__ Wt,
                            const float* __restrict__ asP, const float* __restrict__ adP,
                            int nr,
                            __bf16* __restrict__ hB, float* __restrict__ ss, float* __restrict__ sd){
  const int h    = blockIdx.y;
  const int i0   = blockIdx.x*16;
  const int lane = threadIdx.x & 31;
  const int half = lane >> 4;
  const int mn   = lane & 15;
  v8f acc[TILES];
  #pragma unroll
  for(int t=0;t<TILES;t++){ v8f z = {0.f,0.f,0.f,0.f,0.f,0.f,0.f,0.f}; acc[t]=z; }
  #pragma unroll
  for(int c=0;c<KCH;c++){
    const uint4* xp = (const uint4*)(xbf + (size_t)(i0+mn)*kpadU + c*16 + (half<<2));
    uint4 alo = xp[0], ahi = xp[2];                     // A-frag: two b128 loads
    v8u au = {alo.x,alo.y,alo.z,alo.w, ahi.x,ahi.y,ahi.z,ahi.w};
    v16bf af = __builtin_bit_cast(v16bf, au);
    #pragma unroll
    for(int t=0;t<TILES;t++){
      int col = (t<<4)+mn;
      const uint4* wp = (const uint4*)(Wt + ((size_t)h*64 + col)*32 + c*16 + (half<<2));
      uint4 blo = wp[0], bhi = wp[2];                   // B-frag: two b128 loads
      v8u bu = {blo.x,blo.y,blo.z,blo.w, bhi.x,bhi.y,bhi.z,bhi.w};
      v16bf bf = __builtin_bit_cast(v16bf, bu);
      acc[t] = __builtin_amdgcn_wmma_f32_16x16x32_bf16(false, af, false, bf, (short)0, acc[t], false, false);
    }
  }
  // epilogue: store h (bf16), fused ss = h.a_src, sd = h.a_dst
  float psrc[8], pdst[8];
  #pragma unroll
  for(int r=0;r<8;r++){ psrc[r]=0.f; pdst[r]=0.f; }
  #pragma unroll
  for(int t=0;t<TILES;t++){
    int col = (t<<4)+mn;
    float as = asP[(h<<6)+col];
    float ad = adP[(h<<6)+col];
    #pragma unroll
    for(int r=0;r<8;r++){
      float val = acc[t][r];
      psrc[r] += val*as; pdst[r] += val*ad;
      int row = i0 + r + (half<<3);
      hB[((size_t)h*nr + row)*HID + col] = (__bf16)val;
    }
  }
  #pragma unroll
  for(int r=0;r<8;r++){
    float s=psrc[r], d=pdst[r];
    for(int o=8;o>=1;o>>=1){ s += __shfl_xor(s,o,32); d += __shfl_xor(d,o,32); }
    int row = i0 + r + (half<<3);
    if(mn==0){ ss[(size_t)h*nr+row]=s; sd[(size_t)h*nr+row]=d; }
  }
}

// --------------------- fused flash-style masked-softmax GAT aggregation
template<int TILES>
__global__ void k_gat_attn(const __bf16* __restrict__ hB,
                           const float* __restrict__ ss,
                           const float* __restrict__ sd,
                           const unsigned* __restrict__ Mbits,
                           int n, int nr, int words, int fout, int H,
                           float* __restrict__ out){
  constexpr int NCOL = 16*TILES;
  __shared__ __align__(16) unsigned tileU[64*16];   // [col][k/2] packed bf16 pairs
  __shared__ float ssh[32];
  __shared__ float rowf[16];
  const int h    = blockIdx.y;
  const int i0   = blockIdx.x*16;
  const int lane = threadIdx.x & 31;
  const int half = lane >> 4;
  const int mn   = lane & 15;
  const unsigned short* hU = (const unsigned short*)(hB + (size_t)h*nr*HID);
  const float* ssp = ss + (size_t)h*nr;
  const int row   = i0+mn;
  const float sdv = sd[(size_t)h*nr + row];
  float mrun = -1e30f, lrun = 0.f;
  v8f acc[TILES];
  #pragma unroll
  for(int t=0;t<TILES;t++){ v8f z = {0.f,0.f,0.f,0.f,0.f,0.f,0.f,0.f}; acc[t]=z; }

  const int nch = (n+31)>>5;
  for(int ch=0; ch<nch; ch++){
    const int j0 = ch<<5;
    __syncthreads();
    ssh[lane] = ssp[j0+lane];
    for(int idx=lane; idx<NCOL*16; idx+=32){          // stage h tile, fragment order
      int col = idx & (NCOL-1), vv = idx / NCOL;
      int j = j0 + (vv<<1);
      unsigned lo = (unsigned)hU[(size_t)j*HID + col];
      unsigned hi = (unsigned)hU[(size_t)(j+1)*HID + col];
      tileU[(col<<4)+vv] = lo | (hi<<16);
    }
    __syncthreads();

    unsigned adjw = Mbits[(size_t)row*words + (j0>>5)];
    float sarr[16]; float cmax = -1e30f;
    #pragma unroll
    for(int e=0;e<16;e++){                            // scores in A-fragment layout
      int v=e>>1, p=e&1;
      int k = ((v>>2)<<4) + (half<<3) + ((v&3)<<1) + p;
      float sc = sdv + ssh[k];
      sc = sc>0.f ? sc : 0.2f*sc;                     // leaky_relu
      sc = ((adjw>>k)&1u) ? sc : -1e9f;               // mask
      sarr[e]=sc; cmax = fmaxf(cmax, sc);
    }
    cmax = fmaxf(cmax, __shfl_xor(cmax,16,32));
    float mnew = fmaxf(mrun, cmax);
    float corr = __expf(mrun - mnew);
    mrun = mnew;
    float ps = 0.f;
    #pragma unroll
    for(int e=0;e<16;e++){ float pe = __expf(sarr[e]-mnew); sarr[e]=pe; ps+=pe; }
    ps += __shfl_xor(ps,16,32);
    lrun = lrun*corr + ps;

    __syncthreads();
    if(lane<16) rowf[lane] = corr;
    __syncthreads();
    float cs[8];
    #pragma unroll
    for(int r=0;r<8;r++) cs[r] = rowf[r + (half<<3)];
    #pragma unroll
    for(int t=0;t<TILES;t++){
      #pragma unroll
      for(int r=0;r<8;r++) acc[t][r] *= cs[r];
    }

    v16bf pf;
    #pragma unroll
    for(int e=0;e<16;e++) pf[e] = (__bf16)sarr[e];
    #pragma unroll
    for(int t=0;t<TILES;t++){
      const uint4* tp = (const uint4*)&tileU[(((t<<4)+mn)<<4) + (half<<2)];
      uint4 blo = tp[0], bhi = tp[2];                 // two ds_load_b128
      v8u bu = {blo.x,blo.y,blo.z,blo.w, bhi.x,bhi.y,bhi.z,bhi.w};
      v16bf bf = __builtin_bit_cast(v16bf, bu);
      acc[t] = __builtin_amdgcn_wmma_f32_16x16x32_bf16(false, pf, false, bf, (short)0, acc[t], false, false);
    }
  }

  float inv = 1.f/lrun;
  __syncthreads();
  if(lane<16) rowf[lane] = inv;
  __syncthreads();
  float invH = 1.f/(float)H;
  #pragma unroll
  for(int t=0;t<TILES;t++){
    int col = (t<<4)+mn;
    #pragma unroll
    for(int r=0;r<8;r++){
      int ro = i0 + r + (half<<3);
      if(ro<n && col<fout)
        atomicAdd(&out[(size_t)ro*fout + col], acc[t][r]*rowf[r+(half<<3)]*invH);
    }
  }
}

// ---------------------------------------------------- pooling / unpooling
__global__ void k_score(const float* __restrict__ attn, const float* __restrict__ w,
                        int n, int f, float* __restrict__ score){
  int i = blockIdx.x*blockDim.x+threadIdx.x; if(i>=n) return;
  float nrm=0.f; for(int c=0;c<f;c++) nrm += w[c]*w[c];
  nrm = sqrtf(nrm);
  float dot=0.f; for(int c=0;c<f;c++) dot += attn[(size_t)i*f+c]*w[c];
  score[i] = tanhf(dot/nrm);
}

__global__ void k_topk(const float* __restrict__ s, int n, int k, int* __restrict__ perm){
  int i = blockIdx.x*blockDim.x+threadIdx.x; if(i>=n) return;
  float si = s[i]; int rank = 0;
  for(int j=0;j<n;j++){ float sj = s[j]; rank += (sj>si) || (sj==si && j<i); }
  if(rank<k) perm[rank] = i;
}

__global__ void k_pool_gather(const float* __restrict__ f, const float* __restrict__ score,
                              const int* __restrict__ perm, int k, int fdim, float* __restrict__ o){
  int idx = blockIdx.x*blockDim.x+threadIdx.x; if(idx>=k*fdim) return;
  int r = idx/fdim, c = idx - r*fdim;
  int p = perm[r];
  float v = f[(size_t)p*fdim + c];
  o[idx] = (v>0.f ? v : 0.f) * score[p];
}

__global__ void k_unpool(const float* __restrict__ z, const int* __restrict__ perm,
                         int k, int fdim, float* __restrict__ up){
  int idx = blockIdx.x*blockDim.x+threadIdx.x; if(idx>=k*fdim) return;
  int r = idx/fdim, c = idx - r*fdim;
  up[(size_t)perm[r]*fdim + c] = z[idx];
}

__global__ void k_bits_to_f32(const unsigned* __restrict__ A, int n, int words, float* __restrict__ o){
  int idx = blockIdx.x*blockDim.x+threadIdx.x; if(idx>=n*n) return;
  int i = idx/n, j = idx - i*n;
  o[idx] = ((A[(size_t)i*words + (j>>5)] >> (j&31)) & 1u) ? 1.f : 0.f;
}

// --------------------------------------------------------------- driver
extern "C" void kernel_launch(void* const* d_in, const int* in_sizes, int n_in,
                              void* d_out, int out_size, void* d_ws, size_t ws_size,
                              hipStream_t stream){
  const float* x  = (const float*)d_in[0];
  const int*   ei = (const int*)d_in[1];
  const float* Wd[3]  = {(const float*)d_in[2], (const float*)d_in[5], (const float*)d_in[8]};
  const float* asd[3] = {(const float*)d_in[3], (const float*)d_in[6], (const float*)d_in[9]};
  const float* add_[3]= {(const float*)d_in[4], (const float*)d_in[7], (const float*)d_in[10]};
  const float* Wu[3]  = {(const float*)d_in[11],(const float*)d_in[14],(const float*)d_in[17]};
  const float* asu[3] = {(const float*)d_in[12],(const float*)d_in[15],(const float*)d_in[18]};
  const float* adu[3] = {(const float*)d_in[13],(const float*)d_in[16],(const float*)d_in[19]};
  const float* pw[3]  = {(const float*)d_in[20],(const float*)d_in[21],(const float*)d_in[22]};
  const int nE = in_sizes[1]/2;

  char* ws = (char*)d_ws;
  size_t off = 0;
  auto take = [&](size_t bytes)->char*{ char* p = ws+off; off = (off+bytes+255)&~(size_t)255; return p; };
  unsigned* A[4]; for(int i=0;i<4;i++) A[i] = (unsigned*)take((size_t)NMAX*WMAX*4);
  unsigned* At   = (unsigned*)take((size_t)NMAX*WMAX*4);
  unsigned* Asq  = (unsigned*)take((size_t)NMAX*WMAX*4);
  __bf16* hB     = (__bf16*)take((size_t)NHEAD*NMAX*HID*2);
  float* ss      = (float*)take((size_t)NHEAD*NMAX*4);
  float* sd      = (float*)take((size_t)NHEAD*NMAX*4);
  unsigned* xbf  = (unsigned*)take((size_t)NMAX*32*4);     // padded bf16 features
  unsigned* Wt   = (unsigned*)take((size_t)NHEAD*64*32*4); // padded bf16 W^T
  float* asP     = (float*)take((size_t)NHEAD*64*4);
  float* adP     = (float*)take((size_t)NHEAD*64*4);
  float* attn    = (float*)take((size_t)NMAX*HID*4);
  float* fA      = (float*)take((size_t)NMAX*HID*4);
  float* fB      = (float*)take((size_t)NMAX*HID*4);
  float* up      = (float*)take((size_t)NMAX*HID*4);
  float* score   = (float*)take((size_t)NMAX*4);
  int* perm[3]; for(int i=0;i<3;i++) perm[i] = (int*)take((size_t)NMAX*4);

  const int T = 256;
  auto nb  = [&](int n){ return (n+T-1)/T; };
  auto wrd = [&](int n){ return (n+31)>>5; };
  const int ns[4] = {4096, 3277, 2622, 2098};   // k = ceil(0.8*n)

  // adjacency A0 (forced diag); clear reused bit buffers so padded rows are clean
  k_zero_u32<<<nb(NMAX*WMAX), T, 0, stream>>>(A[0], NMAX*WMAX);
  for(int i=1;i<4;i++) k_zero_u32<<<nb(NMAX*WMAX), T, 0, stream>>>(A[i], NMAX*WMAX);
  k_zero_u32<<<nb(NMAX*WMAX), T, 0, stream>>>(At,  NMAX*WMAX);
  k_build_adj<<<nb(nE), T, 0, stream>>>(ei, nE, A[0], ns[0], WMAX);

  auto run_sgat = [&](const float* fin_p, int nn, int fdim_in,
                      const float* W, const float* as_, const float* ad_,
                      int fdim_out, int H, const unsigned* mask, int wds, float* outp){
    const int kch  = (fdim_in  > 32) ? 2 : 1;
    const int tls  = (fdim_out > 32) ? 4 : 2;
    const int nr16 = (nn+15)&~15, nr32 = (nn+31)&~31;
    const int kpadU = kch*16;
    k_pack_feat<<<nb(nr16*kpadU), T, 0, stream>>>(fin_p, nn, fdim_in, kpadU, xbf, nr16);
    k_pack_w<<<nb(H*64*32), T, 0, stream>>>(W, fdim_in, fdim_out, H, Wt);
    k_pack_vec<<<nb(H*64), T, 0, stream>>>(as_, fdim_out, H, asP);
    k_pack_vec<<<nb(H*64), T, 0, stream>>>(ad_, fdim_out, H, adP);
    k_zero_u32<<<nb(H*nr32*32), T, 0, stream>>>((unsigned*)hB, H*nr32*32);  // 64 bf16 = 32 u32/row
    k_zero_f32<<<nb(H*nr32), T, 0, stream>>>(ss, H*nr32);
    k_zero_f32<<<nb(H*nr32), T, 0, stream>>>(sd, H*nr32);
    dim3 g((nn+15)/16, H);
    if(kch==1)        k_gat_xform<1,4><<<g,32,0,stream>>>(xbf,kpadU,Wt,asP,adP,nr32,hB,ss,sd);
    else if(tls==4)   k_gat_xform<2,4><<<g,32,0,stream>>>(xbf,kpadU,Wt,asP,adP,nr32,hB,ss,sd);
    else              k_gat_xform<2,2><<<g,32,0,stream>>>(xbf,kpadU,Wt,asP,adP,nr32,hB,ss,sd);
    k_zero_f32<<<nb(nn*fdim_out), T, 0, stream>>>(outp, nn*fdim_out);
    if(tls==4) k_gat_attn<4><<<g,32,0,stream>>>(hB,ss,sd,mask,nn,nr32,wds,fdim_out,H,outp);
    else       k_gat_attn<2><<<g,32,0,stream>>>(hB,ss,sd,mask,nn,nr32,wds,fdim_out,H,outp);
  };

  // -------------------------------------------------------------- down path
  const float* fcur = x; int fin_dim = 20;
  float* fnext[3] = {fA, fB, fA};
  for(int L=0; L<3; L++){
    int nn = ns[L], wds = wrd(nn), kk = ns[L+1], wk = wrd(kk);
    k_transpose_bits<<<nb(nn*wds), T, 0, stream>>>(A[L], At, nn, wds);     // M = A^T
    run_sgat(fcur, nn, fin_dim, Wd[L], asd[L], add_[L], HID, NHEAD, At, wds, attn);
    k_sq_adj<<<nn, 128, 0, stream>>>(A[L], Asq, nn, wds);                  // (A@A)>0
    k_score<<<nb(nn), T, 0, stream>>>(attn, pw[L], nn, HID, score);
    k_topk<<<nb(nn), T, 0, stream>>>(score, nn, kk, perm[L]);
    k_pool_gather<<<nb(kk*HID), T, 0, stream>>>(attn, score, perm[L], kk, HID, fnext[L]);
    k_gather_adj<<<nb(kk*wk), T, 0, stream>>>(Asq, perm[L], A[L+1], wds, kk, wk);
    fcur = fnext[L]; fin_dim = HID;
  }
  const float* latent = fcur;   // fA, 2098 x 64
  const float* z = fcur;

  // ---------------------------------------------------------------- up path
  for(int U=0; U<3; U++){
    int idx = 2-U;
    int nbig = ns[idx], wds = wrd(nbig), ksm = ns[idx+1];
    int H  = (U==2) ? 1 : NHEAD;
    int fo = (U==2) ? 20 : HID;
    k_zero_f32<<<nb(nbig*HID), T, 0, stream>>>(up, nbig*HID);
    k_unpool<<<nb(ksm*HID), T, 0, stream>>>(z, perm[idx], ksm, HID, up);
    run_sgat(up, nbig, HID, Wu[U], asu[U], adu[U], fo, H, A[idx], wds, attn);
    if(U<2) k_relu<<<nb(nbig*HID), T, 0, stream>>>(attn, nbig*HID);
    z = attn;
  }

  // ----------------------------------------------------------- outputs
  float* o = (float*)d_out;
  k_copy_f32<<<nb(ns[0]*20), T, 0, stream>>>(z, o, ns[0]*20);
  k_copy_f32<<<nb(ns[3]*HID), T, 0, stream>>>(latent, o + ns[0]*20, ns[3]*HID);
  k_bits_to_f32<<<nb(ns[3]*ns[3]), T, 0, stream>>>(A[3], ns[3], wrd(ns[3]),
                                                   o + ns[0]*20 + ns[3]*HID);
}
